// AudioOnlyGNN_3453153706412
// MI455X (gfx1250) — matrix operand order repaired and checked by reading, not compiled, not run on hardware
//
#include <hip/hip_runtime.h>
#include <hip/hip_bf16.h>

typedef __attribute__((ext_vector_type(16))) __bf16        v16bf;
typedef __attribute__((ext_vector_type(8)))  float         v8f;
typedef __attribute__((ext_vector_type(8)))  unsigned int  v8u;

#define BN_EPS 1e-5f

static __device__ __forceinline__ unsigned short f2bf(float f) {
  unsigned u = __builtin_bit_cast(unsigned, f);
  u += 0x7FFFu + ((u >> 16) & 1u);           // round-to-nearest-even
  return (unsigned short)(u >> 16);
}

static __device__ __forceinline__ void atomAddG(float* p, float v) {
  __hip_atomic_fetch_add(p, v, __ATOMIC_RELAXED, __HIP_MEMORY_SCOPE_AGENT);
}

// ---------------- init: dinv=1 (self loop), stats=0, pool=0 ----------------
__global__ void k_init(float* dinv, float* stats, float* pool, int N) {
  int i = blockIdx.x * 256 + threadIdx.x;
  if (i < N) dinv[i] = 1.0f;
  if (i < 256) stats[i] = 0.0f;
  if (i < 64 * 64 + 64) pool[i] = 0.0f;
}

// ---------------- column sums / sumsq over x[N,128] ----------------
__global__ void k_stats(const float* __restrict__ x, float* stats, int N) {
  __shared__ float ls[256];                  // [0:128) sum, [128:256) sumsq
  int tid = threadIdx.x;
  ls[tid] = 0.0f;
  __syncthreads();
  int base = blockIdx.x * 256;               // 256 rows per block
  for (int i = tid; i < 256 * 128; i += 256) {
    int r = base + (i >> 7);
    if (r < N) {
      float v = x[(size_t)r * 128 + (i & 127)];
      atomicAdd(&ls[i & 127], v);            // ds_add_f32
      atomicAdd(&ls[128 + (i & 127)], v * v);
    }
  }
  __syncthreads();
  atomAddG(&stats[tid], ls[tid]);
}

// ---------------- BN apply -> bf16 ----------------
__global__ void k_bn(const float* __restrict__ x, const float* __restrict__ stats,
                     const float* __restrict__ gamma, const float* __restrict__ beta,
                     unsigned short* __restrict__ xb, int total, float invN) {
  int i = blockIdx.x * 256 + threadIdx.x;
  if (i >= total) return;
  int c = i & 127;
  float mu  = stats[c] * invN;
  float var = stats[128 + c] * invN - mu * mu;
  float v = (x[i] - mu) * rsqrtf(var + BN_EPS) * gamma[c] + beta[c];
  xb[i] = f2bf(v);
}

// ---------------- degree count on dst, then rsqrt in place ----------------
__global__ void k_deg(const int* __restrict__ ei, float* dinv, int E) {
  int e = blockIdx.x * 256 + threadIdx.x;
  if (e < E) atomAddG(&dinv[ei[E + e]], 1.0f);
}
__global__ void k_rsqrt(float* dinv, int N) {
  int i = blockIdx.x * 256 + threadIdx.x;
  if (i < N) dinv[i] = rsqrtf(dinv[i]);
}

// ---------------- WMMA GEMM: Y[N,Hout] = bf16(X[N,128]) @ bf16(W[128,Hout]) ----------------
// grid.x = N/16 M-tiles, 128 threads = 4 waves; wave owns 16x16 tiles nt, nt+4, ...
// W is staged into LDS in WMMA B-fragment order so each fragment = 2x ds_load_b128.
__global__ void k_gemm(const unsigned int* __restrict__ Xu,  // bf16 pairs, row stride 64 uints
                       const float* __restrict__ W,
                       float* __restrict__ Y, int Hout) {
  __shared__ unsigned int sA[16 * 64];           // raw 16x128 bf16 A tile (pair-packed)
  __shared__ unsigned int sB[8 * 4 * 32 * 8];    // B fragments: [nt][kk32][lane][v], 32KB max
  int tid = threadIdx.x;
  int mtile = blockIdx.x;

  // stage A raw (fully coalesced 128B rows)
  for (int i = tid; i < 16 * 64; i += 128)
    sA[i] = Xu[(size_t)mtile * 1024 + i];

  // stage B fragment-packed: read W row-major coalesced, scatter 16-bit halves in LDS.
  // read side needs: bu[v] = pack(W[kk+khB+2v][col], W[kk+khB+2v+1][col]),
  //   khB = (lane&16)?16:0, col = nt*16 + (lane&15)
  unsigned short* sB16 = (unsigned short*)sB;
  for (int i = tid; i < 128 * Hout; i += 128) {
    int k   = i / Hout;                          // 0..127
    int col = i - k * Hout;
    int k2   = k >> 1;
    int kk32 = k2 >> 4;
    int rem  = k2 & 15;
    int lane = ((rem >> 3) << 4) | (col & 15);   // bit4 = K-half, low4 = n
    int v    = rem & 7;
    int nt   = col >> 4;
    int dest = ((nt * 4 + kk32) * 32 + lane) * 8 + v;
    sB16[dest * 2 + (k & 1)] = f2bf(W[i]);
  }
  __syncthreads();

  int wave = tid >> 5, lane = tid & 31;
  int n    = lane & 15;                          // A row / B,C col index
  int kb_a = (lane & 16) ? 8 : 0;                // A K sub-offset (pair units)

  for (int nt = wave; nt * 16 < Hout; nt += 4) {
    v8f acc = {};
    const unsigned int* aBase = sA + n * 64 + kb_a;
    const unsigned int* bBase = sB + ((nt * 4) * 32 + lane) * 8;
#pragma unroll
    for (int kk32 = 0; kk32 < 4; ++kk32) {
      v8u au, bu;
#pragma unroll
      for (int v = 0; v < 8; ++v) {
        au[v] = aBase[kk32 * 16 + v];                         // 8 contiguous uints
        bu[v] = bBase[kk32 * 32 * 8 + v];                     // 8 contiguous uints
      }
      acc = __builtin_amdgcn_wmma_f32_16x16x32_bf16(
          false, __builtin_bit_cast(v16bf, au),
          false, __builtin_bit_cast(v16bf, bu),
          (short)0, acc, false, false);
    }
    int mbase = mtile * 16 + ((lane & 16) ? 8 : 0);
#pragma unroll
    for (int v = 0; v < 8; ++v)
      Y[(size_t)(mbase + v) * Hout + nt * 16 + n] = acc[v];
  }
}

// ---------------- zero fp32 buffer ----------------
__global__ void k_zero(float* p, int total) {
  int i = blockIdx.x * 256 + threadIdx.x;
  if (i < total) p[i] = 0.0f;
}

// ---------------- edge scatter: one wave per edge ----------------
__global__ void k_scatter(const int* __restrict__ ei, const float* __restrict__ dinv,
                          const float* __restrict__ h, float* agg, int E, int H) {
  int gt = blockIdx.x * 256 + threadIdx.x;
  int e = gt >> 5, lane = gt & 31;
  if (e >= E) return;
  int src = ei[e], dst = ei[E + e];
  float coef = dinv[src] * dinv[dst];
  const float* hs = h + (size_t)src * H;
  float* ad = agg + (size_t)dst * H;
  for (int f = lane; f < H; f += 32)
    atomAddG(&ad[f], hs[f] * coef);
}

// ---------------- self-loop + bias + relu; write bf16 (next conv) or f32 in place ----------------
__global__ void k_finish(const float* __restrict__ h, const float* __restrict__ dinv,
                         const float* __restrict__ bias, float* agg,
                         unsigned short* xb, int total, int H, int writeBF) {
  int i = blockIdx.x * 256 + threadIdx.x;
  if (i >= total) return;
  int nrow = i / H, f = i - nrow * H;
  float dv = dinv[nrow];
  float o = agg[i] + h[i] * dv * dv + bias[f];
  o = fmaxf(o, 0.0f);
  if (writeBF) xb[i] = f2bf(o);
  else         agg[i] = o;
}

// ---------------- per-graph mean pool (sums + counts via atomics) ----------------
__global__ void k_pool(const float* __restrict__ h3, const int* __restrict__ batch,
                       float* pool, int N) {
  int i = blockIdx.x * 256 + threadIdx.x;
  if (i >= N * 64) return;
  int nrow = i >> 6, f = i & 63;
  int g = batch[nrow];
  atomAddG(&pool[g * 64 + f], h3[i]);
  if (f == 0) atomAddG(&pool[64 * 64 + g], 1.0f);
}

// ---------------- classifier head: 1 block, 64 threads (one graph each) ----------------
__global__ void k_cls(const float* __restrict__ pool,
                      const float* __restrict__ Wc1, const float* __restrict__ bc1,
                      const float* __restrict__ Wc2, const float* __restrict__ bc2,
                      float* __restrict__ out) {
  __shared__ float sp[64 * 64];
  int g = threadIdx.x;                       // 64 threads
  float cnt = fmaxf(pool[64 * 64 + g], 1.0f);
  for (int f = 0; f < 64; ++f) sp[g * 64 + f] = pool[g * 64 + f] / cnt;
  __syncthreads();
  float z[32];
#pragma unroll
  for (int j = 0; j < 32; ++j) {
    float s = bc1[j];
    for (int f = 0; f < 64; ++f) s += sp[g * 64 + f] * Wc1[f * 32 + j];
    z[j] = fmaxf(s, 0.0f);
  }
#pragma unroll
  for (int c = 0; c < 2; ++c) {
    float s = bc2[c];
    for (int j = 0; j < 32; ++j) s += z[j] * Wc2[j * 2 + c];
    out[g * 2 + c] = s;
  }
}

extern "C" void kernel_launch(void* const* d_in, const int* in_sizes, int n_in,
                              void* d_out, int out_size, void* d_ws, size_t ws_size,
                              hipStream_t stream) {
  const float* x     = (const float*)d_in[0];
  const int*   ei    = (const int*)d_in[1];   // [2,E] flat: row0=src, row1=dst
  const int*   batch = (const int*)d_in[2];
  const float* gamma = (const float*)d_in[3];
  const float* beta  = (const float*)d_in[4];
  const float* W1 = (const float*)d_in[5];  const float* b1 = (const float*)d_in[6];
  const float* W2 = (const float*)d_in[7];  const float* b2 = (const float*)d_in[8];
  const float* W3 = (const float*)d_in[9];  const float* b3 = (const float*)d_in[10];
  const float* Wc1 = (const float*)d_in[11]; const float* bc1 = (const float*)d_in[12];
  const float* Wc2 = (const float*)d_in[13]; const float* bc2 = (const float*)d_in[14];
  float* out = (float*)d_out;

  const int N = in_sizes[2];
  const int E = in_sizes[1] / 2;

  // workspace layout (256B aligned)
  char* ws = (char*)d_ws;
  size_t off = 0;
  auto carve = [&](size_t bytes) -> void* {
    void* p = ws + off;
    off = (off + bytes + 255) & ~(size_t)255;
    return p;
  };
  unsigned short* xb   = (unsigned short*)carve((size_t)N * 128 * 2); // bf16 GEMM input
  float*          buf1 = (float*)carve((size_t)N * 128 * 4);          // gemm out (h)
  float*          agg  = (float*)carve((size_t)N * 128 * 4);          // scatter accumulator
  float*          dinv = (float*)carve((size_t)N * 4);                // deg -> rsqrt(deg)
  float*          stats = (float*)carve(256 * 4);                     // sums | sumsq
  float*          pool  = (float*)carve((64 * 64 + 64) * 4);          // sums | counts
  (void)ws_size; (void)n_in; (void)out_size;

  const int T = 256;
  const int mtiles = N / 16;                 // N = 50000 -> 3125

  k_init  <<<(N + T - 1) / T, T, 0, stream>>>(dinv, stats, pool, N);
  k_stats <<<(N + T - 1) / T, T, 0, stream>>>(x, stats, N);
  k_bn    <<<((N * 128) + T - 1) / T, T, 0, stream>>>(x, stats, gamma, beta, xb, N * 128, 1.0f / (float)N);
  k_deg   <<<(E + T - 1) / T, T, 0, stream>>>(ei, dinv, E);
  k_rsqrt <<<(N + T - 1) / T, T, 0, stream>>>(dinv, N);

  // conv1: 128 -> 128
  k_gemm   <<<mtiles, 128, 0, stream>>>((const unsigned int*)xb, W1, buf1, 128);
  k_zero   <<<((N * 128) + T - 1) / T, T, 0, stream>>>(agg, N * 128);
  k_scatter<<<((E * 32) + T - 1) / T, T, 0, stream>>>(ei, dinv, buf1, agg, E, 128);
  k_finish <<<((N * 128) + T - 1) / T, T, 0, stream>>>(buf1, dinv, b1, agg, xb, N * 128, 128, 1);

  // conv2: 128 -> 128
  k_gemm   <<<mtiles, 128, 0, stream>>>((const unsigned int*)xb, W2, buf1, 128);
  k_zero   <<<((N * 128) + T - 1) / T, T, 0, stream>>>(agg, N * 128);
  k_scatter<<<((E * 32) + T - 1) / T, T, 0, stream>>>(ei, dinv, buf1, agg, E, 128);
  k_finish <<<((N * 128) + T - 1) / T, T, 0, stream>>>(buf1, dinv, b2, agg, xb, N * 128, 128, 1);

  // conv3: 128 -> 64 (output stays f32, in place in agg)
  k_gemm   <<<mtiles, 128, 0, stream>>>((const unsigned int*)xb, W3, buf1, 64);
  k_zero   <<<((N * 64) + T - 1) / T, T, 0, stream>>>(agg, N * 64);
  k_scatter<<<((E * 32) + T - 1) / T, T, 0, stream>>>(ei, dinv, buf1, agg, E, 64);
  k_finish <<<((N * 64) + T - 1) / T, T, 0, stream>>>(buf1, dinv, b3, agg, xb, N * 64, 64, 0);

  // pool + classifier
  k_pool <<<((N * 64) + T - 1) / T, T, 0, stream>>>(agg, batch, pool, N);
  k_cls  <<<1, 64, 0, stream>>>(pool, Wc1, bc1, Wc2, bc2, out);
}
